// MultiHeadAttention_24275155157480
// MI455X (gfx1250) — matrix operand order repaired and checked by reading, not compiled, not run on hardware
//
#include <hip/hip_runtime.h>

// ---------------------------------------------------------------------------
// MHA forward for MI455X (gfx1250, wave32, WMMA).
//   B=2, S=4096, D_MODEL=512, H=8, Dh=64, M = B*S = 8192
// Pipeline:
//   1) gemm_bias_wmma (mode 1): Q = X Wq^T + bq  -> bf16 [B,H,S,Dh]
//   2) gemm_bias_wmma (mode 1): K                -> bf16 [B,H,S,Dh]
//   3) gemm_bias_wmma (mode 2): V                -> bf16 [B,H,Dh,S] (transposed)
//   4) flash_attn_kernel: causal flash attention -> fp32 [B,S,512]
//   5) gemm_bias_wmma (mode 0): out = O Wo^T + bo -> fp32 d_out
// ---------------------------------------------------------------------------

typedef __attribute__((ext_vector_type(16))) __bf16 v16bf;
typedef __attribute__((ext_vector_type(8)))  __bf16 v8bf;
typedef __attribute__((ext_vector_type(8)))  float  v8f;

#define SEQ   4096
#define DM    512
#define NH    8
#define DH    64
#define MROWS 8192           // B * S
#define TM    128            // GEMM block tile M
#define TN    64             // GEMM block tile N
#define NEG_BIG (-3.0e38f)

// ---------------------------------------------------------------------------
// Generic GEMM: C[m,n] = sum_k A[m,k] * W[n,k] + bias[n]
// A: fp32 [MROWS, 512] row-major.  W: fp32 [512, 512] row-major (out, in).
// mode 0: fp32 out [MROWS, 512]
// mode 1: bf16 out [B,H,S,Dh]   (row-major in Dh)
// mode 2: bf16 out [B,H,Dh,S]   (transposed: contiguous in S)
// ---------------------------------------------------------------------------
__global__ __launch_bounds__(256) void gemm_bias_wmma(
    const float* __restrict__ A, const float* __restrict__ W,
    const float* __restrict__ bias, float* __restrict__ outF,
    __bf16* __restrict__ outB, int mode)
{
  __shared__ __align__(64) __bf16 Wl[TN][DM];   // 64 KB weight tile, bf16

  const int tid  = threadIdx.x;
  const int wave = tid >> 5;
  const int lane = tid & 31;
  const int half = lane >> 4;     // which 16-lane half
  const int l16  = lane & 15;
  const int bm   = blockIdx.x * TM;
  const int bn   = blockIdx.y * TN;

  // ---- cooperatively stage W tile (64 x 512) fp32 -> bf16 in LDS ----------
  for (int i = tid; i < (TN * DM) / 4; i += 256) {
    const int idx = i * 4;
    const int r = idx >> 9;           // / 512
    const int c = idx & (DM - 1);     // % 512
    const float4 w4 = *(const float4*)(W + (size_t)(bn + r) * DM + c);
    Wl[r][c + 0] = (__bf16)w4.x;
    Wl[r][c + 1] = (__bf16)w4.y;
    Wl[r][c + 2] = (__bf16)w4.z;
    Wl[r][c + 3] = (__bf16)w4.w;
  }
  __syncthreads();

  // ---- each wave computes a 16 x 64 strip of C ----------------------------
  const float* arow = A + (size_t)(bm + wave * 16 + l16) * DM;

  const v8f zero = {0.f, 0.f, 0.f, 0.f, 0.f, 0.f, 0.f, 0.f};
  v8f acc[4] = {zero, zero, zero, zero};

  for (int kb = 0; kb < DM; kb += 32) {
    // A fragment: lane holds row (l16); 16 consecutive K per lane half
    const int o1 = kb + half * 16;
    const int o2 = kb + half * 16 + 8;
    float tmp[16];
    *(float4*)(tmp + 0)  = *(const float4*)(arow + o1);
    *(float4*)(tmp + 4)  = *(const float4*)(arow + o1 + 4);
    *(float4*)(tmp + 8)  = *(const float4*)(arow + o2);
    *(float4*)(tmp + 12) = *(const float4*)(arow + o2 + 4);
    v16bf af;
#pragma unroll
    for (int j = 0; j < 16; ++j) af[j] = (__bf16)tmp[j];

#pragma unroll
    for (int t = 0; t < 4; ++t) {
      // B fragment: lane holds column n = t*16+l16, 16 consecutive K from LDS
      const v16bf bf = *(const v16bf*)&Wl[t * 16 + l16][kb + half * 16];
      acc[t] = __builtin_amdgcn_wmma_f32_16x16x32_bf16(
          false, af, false, bf, (short)0, acc[t], false, false);
    }
  }

  // ---- epilogue: C layout lane=col(l16), vgpr r = row r + 8*half ----------
#pragma unroll
  for (int t = 0; t < 4; ++t) {
    const int n  = bn + t * 16 + l16;
    const float bv = bias[n];
#pragma unroll
    for (int r = 0; r < 8; ++r) {
      const int m   = bm + wave * 16 + r + half * 8;
      const float v = acc[t][r] + bv;
      if (mode == 0) {
        outF[(size_t)m * DM + n] = v;
      } else {
        const int b = m >> 12, s = m & (SEQ - 1);
        const int h = n >> 6, d = n & (DH - 1);
        const size_t idx = (mode == 1)
            ? ((size_t)(b * NH + h) * SEQ + s) * DH + d
            : ((size_t)(b * NH + h) * DH + d) * SEQ + s;
        outB[idx] = (__bf16)v;
      }
    }
  }
}

// ---------------------------------------------------------------------------
// Causal flash attention.  grid = (S/128, B*H), block = 256 (8 waves).
// Each wave owns 16 query rows; loops over 64-key tiles up to the diagonal.
// Qb/Kb: bf16 [B,H,S,Dh].  Vt: bf16 [B,H,Dh,S].  O: fp32 [B,S,512].
// ---------------------------------------------------------------------------
__global__ __launch_bounds__(256) void flash_attn_kernel(
    const __bf16* __restrict__ Qb, const __bf16* __restrict__ Kb,
    const __bf16* __restrict__ Vt, float* __restrict__ O)
{
  __shared__ __align__(64) __bf16 Pl[8][16][DH];   // per-wave P transpose, 16 KB

  const int tid  = threadIdx.x;
  const int wave = tid >> 5;
  const int lane = tid & 31;
  const int half = lane >> 4;
  const int l16  = lane & 15;
  const int bh   = blockIdx.y;                       // b*NH + h
  const int qr0  = blockIdx.x * 128 + wave * 16;     // first query row of wave
  const size_t qkBase = (size_t)bh * SEQ * DH;
  const size_t vBase  = (size_t)bh * DH * SEQ;

  // ---- Q fragments (K = Dh = 64 -> two 16x32 A fragments), kept resident --
  v16bf qf[2];
  {
    const __bf16* qrow = Qb + qkBase + (size_t)(qr0 + l16) * DH;
#pragma unroll
    for (int c = 0; c < 2; ++c) {
      const v8bf q0 = *(const v8bf*)(qrow + c * 32 + half * 16);
      const v8bf q1 = *(const v8bf*)(qrow + c * 32 + half * 16 + 8);
#pragma unroll
      for (int j = 0; j < 8; ++j) { qf[c][j] = q0[j]; qf[c][8 + j] = q1[j]; }
    }
  }

  const v8f zero = {0.f, 0.f, 0.f, 0.f, 0.f, 0.f, 0.f, 0.f};
  v8f o[4] = {zero, zero, zero, zero};
  float mrun[8], lrun[8];
#pragma unroll
  for (int r = 0; r < 8; ++r) { mrun[r] = NEG_BIG; lrun[r] = 0.f; }

  const int kvEnd = qr0 + 16;                        // causal bound (exclusive)
  for (int kv0 = 0; kv0 < kvEnd; kv0 += 64) {
    // ---- S = Q K^T (scaled later): 4 col-tiles x 2 K-chunks = 8 WMMA ------
    v8f s[4];
#pragma unroll
    for (int t = 0; t < 4; ++t) {
      v8f z = zero;
#pragma unroll
      for (int c = 0; c < 2; ++c) {
        const __bf16* kp = Kb + qkBase +
            (size_t)(kv0 + t * 16 + l16) * DH + c * 32 + half * 16;
        const v16bf kf = *(const v16bf*)kp;
        z = __builtin_amdgcn_wmma_f32_16x16x32_bf16(
            false, qf[c], false, kf, (short)0, z, false, false);
      }
      s[t] = z;
    }

    // ---- scale 1/sqrt(Dh) + causal mask -----------------------------------
#pragma unroll
    for (int t = 0; t < 4; ++t) {
      const int kj = kv0 + t * 16 + l16;
#pragma unroll
      for (int r = 0; r < 8; ++r) {
        const int qi = qr0 + r + half * 8;
        const float v = s[t][r] * 0.125f;
        s[t][r] = (kj > qi) ? NEG_BIG : v;
      }
    }

    // ---- online softmax (row stats live per-lane across 16-lane half) -----
#pragma unroll
    for (int r = 0; r < 8; ++r) {
      float mx = s[0][r];
#pragma unroll
      for (int t = 1; t < 4; ++t) mx = fmaxf(mx, s[t][r]);
#pragma unroll
      for (int dd = 1; dd < 16; dd <<= 1) mx = fmaxf(mx, __shfl_xor(mx, dd, 32));
      const float mnew  = fmaxf(mrun[r], mx);
      const float alpha = __expf(mrun[r] - mnew);
      float rs = 0.f;
#pragma unroll
      for (int t = 0; t < 4; ++t) {
        const float p = __expf(s[t][r] - mnew);
        s[t][r] = p;
        rs += p;
      }
#pragma unroll
      for (int dd = 1; dd < 16; dd <<= 1) rs += __shfl_xor(rs, dd, 32);
      lrun[r] = lrun[r] * alpha + rs;
      mrun[r] = mnew;
#pragma unroll
      for (int t = 0; t < 4; ++t) o[t][r] *= alpha;
    }

    // ---- transpose P: C-layout -> A-layout via per-wave LDS region --------
#pragma unroll
    for (int t = 0; t < 4; ++t)
#pragma unroll
      for (int r = 0; r < 8; ++r)
        Pl[wave][r + half * 8][t * 16 + l16] = (__bf16)s[t][r];

    asm volatile("s_wait_dscnt 0" ::: "memory");   // wave-private region; no barrier

    v16bf pf[2];
#pragma unroll
    for (int c = 0; c < 2; ++c) {
      const v8bf p0 = *(const v8bf*)&Pl[wave][l16][c * 32 + half * 16];
      const v8bf p1 = *(const v8bf*)&Pl[wave][l16][c * 32 + half * 16 + 8];
#pragma unroll
      for (int j = 0; j < 8; ++j) { pf[c][j] = p0[j]; pf[c][8 + j] = p1[j]; }
    }

    // ---- O += P V : V^T stored so per-lane key-runs are contiguous --------
#pragma unroll
    for (int t = 0; t < 4; ++t) {
#pragma unroll
      for (int c = 0; c < 2; ++c) {
        const __bf16* vp = Vt + vBase + (size_t)(t * 16 + l16) * SEQ +
                           kv0 + c * 32 + half * 16;
        const v16bf vf = *(const v16bf*)vp;
        o[t] = __builtin_amdgcn_wmma_f32_16x16x32_bf16(
            false, pf[c], false, vf, (short)0, o[t], false, false);
      }
    }
  }

  // ---- normalize and write O (fp32, [B,S,512] with head-interleave) -------
  const int b = bh >> 3, h = bh & 7;
#pragma unroll
  for (int r = 0; r < 8; ++r) {
    const int qrow  = qr0 + r + half * 8;
    const float inv = 1.0f / lrun[r];
#pragma unroll
    for (int t = 0; t < 4; ++t) {
      const int d = t * 16 + l16;
      O[((size_t)b * SEQ + qrow) * DM + h * DH + d] = o[t][r] * inv;
    }
  }
}

// ---------------------------------------------------------------------------
extern "C" void kernel_launch(void* const* d_in, const int* in_sizes, int n_in,
                              void* d_out, int out_size, void* d_ws, size_t ws_size,
                              hipStream_t stream) {
  (void)in_sizes; (void)n_in; (void)out_size; (void)ws_size;

  const float* Xq = (const float*)d_in[0];
  const float* Xk = (const float*)d_in[1];
  const float* Xv = (const float*)d_in[2];
  const float* Wq = (const float*)d_in[3];
  const float* bq = (const float*)d_in[4];
  const float* Wk = (const float*)d_in[5];
  const float* bk = (const float*)d_in[6];
  const float* Wv = (const float*)d_in[7];
  const float* bv = (const float*)d_in[8];
  const float* Wo = (const float*)d_in[9];
  const float* bo = (const float*)d_in[10];

  char* ws = (char*)d_ws;
  __bf16* Qb  = (__bf16*)(ws);                      //  8 MiB bf16 [B,H,S,Dh]
  __bf16* Kb  = (__bf16*)(ws + (8u << 20));         //  8 MiB bf16 [B,H,S,Dh]
  __bf16* Vt  = (__bf16*)(ws + (16u << 20));        //  8 MiB bf16 [B,H,Dh,S]
  float*  Oat = (float*)(ws + (24u << 20));         // 16 MiB fp32 [B,S,512]

  const dim3 gGemm(MROWS / TM, DM / TN);            // 64 x 8
  const dim3 blk(256);

  gemm_bias_wmma<<<gGemm, blk, 0, stream>>>(Xq, Wq, bq, nullptr, Qb, 1);
  gemm_bias_wmma<<<gGemm, blk, 0, stream>>>(Xk, Wk, bk, nullptr, Kb, 1);
  gemm_bias_wmma<<<gGemm, blk, 0, stream>>>(Xv, Wv, bv, nullptr, Vt, 2);

  flash_attn_kernel<<<dim3(SEQ / 128, 2 * NH), blk, 0, stream>>>(Qb, Kb, Vt, Oat);

  gemm_bias_wmma<<<gGemm, blk, 0, stream>>>(Oat, Wo, bo, (float*)d_out, nullptr, 0);
}
